// ReconBlock_14070312862112
// MI455X (gfx1250) — compile-verified, matrix operand order checked by reading, not run on hardware
//
#include <hip/hip_runtime.h>

typedef __attribute__((ext_vector_type(16))) _Float16 v16h;
typedef __attribute__((ext_vector_type(8)))  _Float16 v8h;
typedef __attribute__((ext_vector_type(8)))  float    v8f;
typedef __attribute__((ext_vector_type(4)))  float    v4f;

#define C_CH        256
#define M_TILE      64             // rows per block
#define N_TILE      128            // cols per block (2 blocks cover C=256)
#define K_CHUNKS    24             // 768 / 32 (global, concat over 3 taps)
#define KK_PER_TAP  8              // 256 / 32
#define NFRAG_TOT   16             // 256 / 16
#define FRAG_HALVES 512            // 32 lanes * 16 halves per B fragment
#define LDS_STRIDE  264            // 256 + 8 halves pad -> conflict-free frag reads
#define WS_FEATS_OFF (1u << 21)    // feats16 at +2 MB in workspace (weights below)

// ---------------------------------------------------------------------------
// Prep 1: repack fp32 weights [axis][tap][Cin][Cout] -> f16 WMMA B-fragment
// order (16-bit operand layout, CDNA5 ISA):
//   lanes 0-15 : N = lane,    halves e=0..7 -> K=e,    e=8..15 -> K=e+8
//   lanes 16-31: N = lane-16, halves e=0..7 -> K=e+8,  e=8..15 -> K=e+16
// ---------------------------------------------------------------------------
__global__ void pack_weights(const float* __restrict__ w1,
                             const float* __restrict__ w2,
                             const float* __restrict__ w3,
                             _Float16* __restrict__ wsB) {
    int idx = blockIdx.x * blockDim.x + threadIdx.x;
    const int total = 3 * K_CHUNKS * NFRAG_TOT * FRAG_HALVES;
    if (idx >= total) return;
    int e  = idx & 15;
    int l  = (idx >> 4) & 31;
    int nf = (idx >> 9) & 15;
    int kk = (idx >> 13) % K_CHUNKS;
    int a  = idx / (FRAG_HALVES * NFRAG_TOT * K_CHUNKS);
    int klocal = (e < 8 ? e : e + 8) + ((l < 16) ? 0 : 8);
    int kg = kk * 32 + klocal;          // global K in 0..767
    int t  = kg >> 8;                   // tap 0..2
    int c  = kg & 255;                  // input channel
    int n  = nf * 16 + (l & 15);        // output channel
    const float* w = (a == 0) ? w1 : (a == 1) ? w2 : w3;
    wsB[idx] = (_Float16)w[(t * C_CH + c) * C_CH + n];
}

// ---------------------------------------------------------------------------
// Prep 2: feats fp32 -> f16 copy (one conversion per element instead of
// re-converting on each of the 9 gathers).
// ---------------------------------------------------------------------------
__global__ void cvt_feats(const float* __restrict__ feats,
                          _Float16* __restrict__ feats16, int total8) {
    int i = blockIdx.x * blockDim.x + threadIdx.x;
    if (i >= total8) return;
    const float* src = feats + (size_t)i * 8;
    v4f a = *(const v4f*)(src);
    v4f b = *(const v4f*)(src + 4);
    v8h h;
    h[0] = (_Float16)a.x; h[1] = (_Float16)a.y;
    h[2] = (_Float16)a.z; h[3] = (_Float16)a.w;
    h[4] = (_Float16)b.x; h[5] = (_Float16)b.y;
    h[6] = (_Float16)b.z; h[7] = (_Float16)b.w;
    *(v8h*)(feats16 + (size_t)i * 8) = h;
}

// ---------------------------------------------------------------------------
// Fused main kernel.  Grid: (row tiles, 2 column halves).
// Block = 64 rows x 128 cols; 8 wave32s tile 2(M: 2 frags) x 4(N: 2 frags).
// 9 (axis,tap) stages fully unrolled: flush prefetched tap tile regs -> LDS,
// prefetch next tap into registers (hidden under WMMAs), 8-step K GEMM with
// 4 WMMAs per step, per-axis BN+sigmoid epilogue, final residual multiply.
// ---------------------------------------------------------------------------
__global__ __launch_bounds__(256)
void recon_block(const float* __restrict__ feats,
                 const _Float16* __restrict__ feats16,
                 const _Float16* __restrict__ wsB,
                 const int* __restrict__ nbr0,
                 const int* __restrict__ nbr1,
                 const int* __restrict__ nbr2,
                 const float* __restrict__ g0, const float* __restrict__ b0,
                 const float* __restrict__ g1, const float* __restrict__ b1,
                 const float* __restrict__ g2, const float* __restrict__ b2,
                 float* __restrict__ out, int N)
{
    __shared__ _Float16 lds[M_TILE * LDS_STRIDE];   // 33792 B

    const int tid  = threadIdx.x;
    const int lane = tid & 31;
    const int wave = tid >> 5;
    const int mi   = wave >> 2;      // 0..1 : rows 32*mi .. 32*mi+31 (2 frags)
    const int ng   = wave & 3;       // 0..3 : cols 32*ng .. 32*ng+31 (2 frags)
    const int row0 = blockIdx.x * M_TILE;
    const int ch   = blockIdx.y;     // column half 0/1
    const int gcol = lane * 8;       // this lane's 16B gather column slice

    const float bnscale = rsqrtf(1.0f + 1e-5f);

    float sum[2][2][8];
    #pragma unroll
    for (int f = 0; f < 2; ++f)
        #pragma unroll
        for (int nf = 0; nf < 2; ++nf)
            #pragma unroll
            for (int r = 0; r < 8; ++r) sum[f][nf][r] = 0.0f;

    // A-fragment addressing (16-bit A layout): row = lane%16 within frag,
    // halves cover K {0..7,16..23} (lanes<16) / {8..15,24..31} (lanes>=16)
    const int arow = 32 * mi + (lane & 15);
    const int asel = (lane < 16) ? 0 : 8;
    const int nfg0 = ch * 8 + ng * 2;    // first of this wave's 2 B frags

    v8h pf[8];      // prefetched tap tile (this thread's 8 chunks)
    int pg[8];      // neighbor index per chunk (<0 => zero fill)

    // ---- branchless prefetch of one tap's tile into registers
    auto prefetch = [&](const int* __restrict__ nbr, int t) {
        #pragma unroll
        for (int i = 0; i < 8; ++i) {
            int grow  = row0 + i * 8 + wave;
            int rcl   = (grow < N) ? grow : (N - 1);
            int g     = nbr[rcl * 3 + t];              // always issued
            g         = (grow < N) ? g : -1;
            pg[i]     = g;
            size_t gi = (size_t)((g >= 0) ? g : 0);
            pf[i]     = *(const v8h*)(feats16 + gi * C_CH + gcol);
        }
    };

    prefetch(nbr0, 0);

    v8f acc[2][2];

    #pragma unroll
    for (int s = 0; s < 9; ++s) {
        const int a = s / 3;         // compile-time after unroll
        const int t = s % 3;

        if (t == 0) {
            #pragma unroll
            for (int f = 0; f < 2; ++f)
                #pragma unroll
                for (int nf = 0; nf < 2; ++nf) acc[f][nf] = (v8f){0,0,0,0,0,0,0,0};
        }

        __syncthreads();   // previous stage done reading LDS
        // ---- flush prefetched tile to LDS (zero-fill via cndmask)
        #pragma unroll
        for (int i = 0; i < 8; ++i) {
            int row = i * 8 + wave;
            v8h h = (pg[i] >= 0) ? pf[i] : (v8h){0,0,0,0,0,0,0,0};
            *(v8h*)(&lds[row * LDS_STRIDE + gcol]) = h;
        }
        __syncthreads();

        // ---- prefetch next stage (hidden under this stage's WMMAs)
        if (s < 8) {
            const int an = (s + 1) / 3;
            const int tn = (s + 1) % 3;
            const int* __restrict__ nxt = (an == 0) ? nbr0 : (an == 1) ? nbr1 : nbr2;
            prefetch(nxt, tn);
        }

        // ---- GEMM: 4 WMMAs (2 M-frags x 2 N-frags) per 32-wide K step
        const _Float16* __restrict__ wbase =
            wsB + (((size_t)a * K_CHUNKS + t * KK_PER_TAP) * NFRAG_TOT) * FRAG_HALVES;

        #pragma unroll 1
        for (int kk = 0; kk < KK_PER_TAP; ++kk) {
            v16h A[2];
            #pragma unroll
            for (int f = 0; f < 2; ++f) {
                const _Float16* ap =
                    &lds[(arow + 16 * f) * LDS_STRIDE + kk * 32 + asel];
                v8h p0 = *(const v8h*)(ap);        // ds_load_b128
                v8h p1 = *(const v8h*)(ap + 16);   // ds_load_b128
                #pragma unroll
                for (int j = 0; j < 8; ++j) { A[f][j] = p0[j]; A[f][8 + j] = p1[j]; }
            }
            const _Float16* bp =
                wbase + ((size_t)kk * NFRAG_TOT + nfg0) * FRAG_HALVES + lane * 16;
            #pragma unroll
            for (int nf = 0; nf < 2; ++nf) {
                v16h B = *(const v16h*)(bp + nf * FRAG_HALVES);  // 32B/lane
                #pragma unroll
                for (int f = 0; f < 2; ++f) {
                    acc[f][nf] = __builtin_amdgcn_wmma_f32_16x16x32_f16(
                        false, A[f], false, B, (short)0, acc[f][nf], false, false);
                }
            }
        }

        // ---- per-axis epilogue: BN (eval) + sigmoid, accumulate
        if (t == 2) {
            const float* __restrict__ gg = (a == 0) ? g0 : (a == 1) ? g1 : g2;
            const float* __restrict__ bb = (a == 0) ? b0 : (a == 1) ? b1 : b2;
            #pragma unroll
            for (int nf = 0; nf < 2; ++nf) {
                int col  = ch * N_TILE + ng * 32 + nf * 16 + (lane & 15);
                float sc = gg[col] * bnscale;
                float bi = bb[col];
                #pragma unroll
                for (int f = 0; f < 2; ++f)
                    #pragma unroll
                    for (int r = 0; r < 8; ++r) {
                        float x = acc[f][nf][r] * sc + bi;
                        sum[f][nf][r] += 1.0f / (1.0f + __expf(-x));
                    }
            }
        }
    }

    // ---- residual multiply + store (C/D layout: lane%16 = col, vgpr r = row)
    #pragma unroll
    for (int nf = 0; nf < 2; ++nf) {
        int col = ch * N_TILE + ng * 32 + nf * 16 + (lane & 15);
        #pragma unroll
        for (int f = 0; f < 2; ++f)
            #pragma unroll
            for (int r = 0; r < 8; ++r) {
                int row = row0 + 32 * mi + 16 * f + ((lane < 16) ? r : r + 8);
                if (row < N) {
                    size_t off = (size_t)row * C_CH + col;
                    out[off] = sum[f][nf][r] * feats[off];
                }
            }
    }
}

// ---------------------------------------------------------------------------
extern "C" void kernel_launch(void* const* d_in, const int* in_sizes, int n_in,
                              void* d_out, int out_size, void* d_ws, size_t ws_size,
                              hipStream_t stream) {
    const float* feats = (const float*)d_in[0];
    const float* w1    = (const float*)d_in[1];
    const float* w2    = (const float*)d_in[2];
    const float* w3    = (const float*)d_in[3];
    const float* g1    = (const float*)d_in[4];
    const float* b1    = (const float*)d_in[5];
    const float* g2    = (const float*)d_in[6];
    const float* b2    = (const float*)d_in[7];
    const float* g3    = (const float*)d_in[8];
    const float* b3    = (const float*)d_in[9];
    const int*   nbr_x = (const int*)d_in[10];
    const int*   nbr_y = (const int*)d_in[11];
    const int*   nbr_z = (const int*)d_in[12];
    float*       out   = (float*)d_out;

    const int N = in_sizes[0] / C_CH;   // 150000

    _Float16* wsB     = (_Float16*)d_ws;                          // 1.125 MB
    _Float16* feats16 = (_Float16*)((char*)d_ws + WS_FEATS_OFF);  // 76.8 MB

    const int totalW = 3 * K_CHUNKS * NFRAG_TOT * FRAG_HALVES;
    pack_weights<<<(totalW + 255) / 256, 256, 0, stream>>>(w1, w2, w3, wsB);

    const int total8 = (N * C_CH) / 8;
    cvt_feats<<<(total8 + 255) / 256, 256, 0, stream>>>(feats, feats16, total8);

    dim3 grid((N + M_TILE - 1) / M_TILE, C_CH / N_TILE);
    recon_block<<<grid, 256, 0, stream>>>(
        feats, feats16, wsB, nbr_x, nbr_y, nbr_z,
        g1, b1, g2, b2, g3, b3, out, N);
}
